// PromptGetter_55104430407806
// MI455X (gfx1250) — matrix operand order repaired and checked by reading, not compiled, not run on hardware
//
#include <hip/hip_runtime.h>
#include <hip/hip_bf16.h>

typedef __attribute__((ext_vector_type(16))) _Float16 v16h;
typedef __attribute__((ext_vector_type(8)))  float    v8f;

#define IMG      1024
#define SIMRES   64
#define NCELL    16      // cells per dim (1024/64)
#define CELLPIX  64
#define NREF     32
#define NCH      256
#define NPIX     4096    // 64*64
#define NTILE    256     // NPIX/16 pixel tiles
#define THRESH   0.65f
#define FEPS     1e-12f

// ---------------------------------------------------------------------------
// Kernel 1: normalize reference feats (32 x 256) and pack into WMMA A-matrix
// fragment layout (f16, 16x32 tiles): lane L holds row M=L%16, group g=L/16,
// half h -> K = g*16+h.  refA[((mt*8+s)*32+lane)*16+h].
// ---------------------------------------------------------------------------
__global__ void k_ref_pack(const float* __restrict__ ref, _Float16* __restrict__ refA) {
    const int m = blockIdx.x;           // 32 blocks x 64 threads
    const int tid = threadIdx.x;
    const float* row = ref + m * NCH;   // reference_feats [32,1,256]
    __shared__ float red[64];
    float s = 0.0f;
    for (int c = tid; c < NCH; c += 64) { float v = row[c]; s += v * v; }
    red[tid] = s;
    __syncthreads();
    for (int off = 32; off > 0; off >>= 1) {
        if (tid < off) red[tid] += red[tid + off];
        __syncthreads();
    }
    const float inv = 1.0f / (sqrtf(red[0]) + FEPS);
    const int mt = m >> 4, M = m & 15;
    for (int idx = tid; idx < 256; idx += 64) {
        const int g = idx >> 7;             // lane group
        const int rem = idx & 127;
        const int st = rem >> 4;            // k-step 0..7
        const int h  = rem & 15;            // half slot
        const int K  = g * 16 + h;
        const int c  = st * 32 + K;
        refA[(((mt * 8 + st) * 32) + (g * 16 + M)) * 16 + h] = (_Float16)(row[c] * inv);
    }
}

// ---------------------------------------------------------------------------
// Kernel 2: per-pixel L2 normalization of image feats + pack into WMMA
// B-matrix fragment layout (f16, 32x16 tiles): lane L holds col N=L%16,
// group g=L/16, half h -> K = g*16+h.  featB[((t*8+s)*32+lane)*16+h].
// ---------------------------------------------------------------------------
__global__ void k_feat_pack(const float* __restrict__ feat, _Float16* __restrict__ featB) {
    const int p = blockIdx.x * blockDim.x + threadIdx.x;  // 16 x 256 = 4096
    float s = 0.0f;
    for (int c = 0; c < NCH; ++c) { float v = feat[c * NPIX + p]; s += v * v; }
    const float inv = 1.0f / (sqrtf(s) + FEPS);
    const int t = p >> 4, n = p & 15;
    for (int st = 0; st < 8; ++st) {
        for (int g = 0; g < 2; ++g) {
            v16h pack;
            #pragma unroll
            for (int h = 0; h < 16; ++h) {
                const int c = st * 32 + g * 16 + h;   // K = g*16+h within step
                pack[h] = (_Float16)(feat[c * NPIX + p] * inv);
            }
            *reinterpret_cast<v16h*>(featB + (((t * 8 + st) * 32) + (g * 16 + n)) * 16) = pack;
        }
    }
}

// ---------------------------------------------------------------------------
// Kernel 3: cosine-sim GEMM 32x4096x256 via v_wmma_f32_16x16x32_f16.
// One wave per 16-pixel tile; two 16x16 accumulators (refs 0-15, 16-31)
// sharing the B fragment; K=256 -> 8 WMMA steps.
// ---------------------------------------------------------------------------
__global__ void k_gemm(const _Float16* __restrict__ refA,
                       const _Float16* __restrict__ featB,
                       float* __restrict__ sim) {
    const int gtid = blockIdx.x * blockDim.x + threadIdx.x;  // 64 blocks x 128
    const int t    = gtid >> 5;       // wave id == pixel tile (0..255)
    const int lane = threadIdx.x & 31;
    v8f acc0 = {};
    v8f acc1 = {};
    #pragma unroll
    for (int s = 0; s < 8; ++s) {
        const v16h a0 = *reinterpret_cast<const v16h*>(refA + ((s) * 32 + lane) * 16);
        const v16h a1 = *reinterpret_cast<const v16h*>(refA + ((8 + s) * 32 + lane) * 16);
        const v16h b  = *reinterpret_cast<const v16h*>(featB + ((t * 8 + s) * 32 + lane) * 16);
        acc0 = __builtin_amdgcn_wmma_f32_16x16x32_f16(false, a0, false, b,
                                                      (short)0, acc0, false, false);
        acc1 = __builtin_amdgcn_wmma_f32_16x16x32_f16(false, a1, false, b,
                                                      (short)0, acc1, false, false);
    }
    // C/D layout: VGPR r -> M = r + 8*(lane/16), N = lane%16
    const int n = lane & 15, hi = lane >> 4;
    const int p = t * 16 + n;
    #pragma unroll
    for (int r = 0; r < 8; ++r) {
        const int m = r + hi * 8;
        sim[m * NPIX + p]        = acc0[r];
        sim[(16 + m) * NPIX + p] = acc1[r];
    }
}

// ---------------------------------------------------------------------------
// Kernel 4: per (map, cell) bilinear-upsampled max/argmax + min/argmin.
// Cell needs only a 6x6 sim neighborhood; never materialize the 1024x1024 map.
// Grid: 32*256 blocks x 256 threads; each thread evaluates 16 samples.
// ---------------------------------------------------------------------------
__global__ void k_cells(const float* __restrict__ sim,
                        float* __restrict__ cellS, float* __restrict__ cellX,
                        float* __restrict__ cellY,
                        float* __restrict__ cellMinV, int* __restrict__ cellMinI) {
    const int b    = blockIdx.x;
    const int m    = b >> 8;
    const int cell = b & 255;
    const int cy = cell >> 4, cx = cell & 15;
    const int tid = threadIdx.x;

    __shared__ float tile[6][6];
    if (tid < 36) {
        const int r = tid / 6, c = tid % 6;
        int gy = cy * 4 - 1 + r; gy = gy < 0 ? 0 : (gy > 63 ? 63 : gy);
        int gx = cx * 4 - 1 + c; gx = gx < 0 ? 0 : (gx > 63 ? 63 : gx);
        tile[r][c] = sim[m * NPIX + gy * SIMRES + gx];
    }
    __syncthreads();

    const int by = cy * 4 - 1, bx = cx * 4 - 1;
    float bestMax = -3.0e38f; int bestMaxQ = 0x7fffffff;
    float bestMin =  3.0e38f; int bestMinF = 0x7fffffff;
    for (int j = 0; j < 16; ++j) {
        const int q  = tid + (j << 8);
        const int iy = q >> 6, ix = q & 63;
        const int Y = cy * CELLPIX + iy, X = cx * CELLPIX + ix;
        float fy = (Y + 0.5f) * 0.0625f - 0.5f; fy = fminf(fmaxf(fy, 0.0f), 63.0f);
        float fx = (X + 0.5f) * 0.0625f - 0.5f; fx = fminf(fmaxf(fx, 0.0f), 63.0f);
        const int y0 = (int)fy, x0 = (int)fx;
        const float wy = fy - (float)y0, wx = fx - (float)x0;
        const int y1 = (y0 + 1 > 63) ? 63 : y0 + 1;
        const int x1 = (x0 + 1 > 63) ? 63 : x0 + 1;
        const int ly0 = y0 - by, ly1 = y1 - by, lx0 = x0 - bx, lx1 = x1 - bx;
        const float v00 = tile[ly0][lx0], v01 = tile[ly0][lx1];
        const float v10 = tile[ly1][lx0], v11 = tile[ly1][lx1];
        const float v = (1.0f - wy) * ((1.0f - wx) * v00 + wx * v01)
                      +          wy * ((1.0f - wx) * v10 + wx * v11);
        if (v > bestMax || (v == bestMax && q < bestMaxQ)) { bestMax = v; bestMaxQ = q; }
        const int f = Y * IMG + X;
        if (v < bestMin || (v == bestMin && f < bestMinF)) { bestMin = v; bestMinF = f; }
    }

    __shared__ float sMaxV[256]; __shared__ int sMaxI[256];
    __shared__ float sMinV[256]; __shared__ int sMinI[256];
    sMaxV[tid] = bestMax; sMaxI[tid] = bestMaxQ;
    sMinV[tid] = bestMin; sMinI[tid] = bestMinF;
    __syncthreads();
    for (int off = 128; off > 0; off >>= 1) {
        if (tid < off) {
            const float v2 = sMaxV[tid + off]; const int i2 = sMaxI[tid + off];
            if (v2 > sMaxV[tid] || (v2 == sMaxV[tid] && i2 < sMaxI[tid])) {
                sMaxV[tid] = v2; sMaxI[tid] = i2;
            }
            const float w2 = sMinV[tid + off]; const int j2 = sMinI[tid + off];
            if (w2 < sMinV[tid] || (w2 == sMinV[tid] && j2 < sMinI[tid])) {
                sMinV[tid] = w2; sMinI[tid] = j2;
            }
        }
        __syncthreads();
    }
    if (tid == 0) {
        const int q = sMaxI[0];
        cellS[m * 256 + cell] = sMaxV[0];
        cellX[m * 256 + cell] = (float)(cx * CELLPIX + (q & 63));
        cellY[m * 256 + cell] = (float)(cy * CELLPIX + (q >> 6));
        cellMinV[m * 256 + cell] = sMinV[0];
        cellMinI[m * 256 + cell] = sMinI[0];
    }
}

// ---------------------------------------------------------------------------
// Kernel 5: per map — bg argmin across cells, threshold/mask, stable
// descending bitonic sort on (score desc, cell-id asc), write outputs.
// out = [32*256*3 point rows] ++ [32*2 bg coords].
// ---------------------------------------------------------------------------
__global__ void k_final(const float* __restrict__ cellS, const float* __restrict__ cellX,
                        const float* __restrict__ cellY,
                        const float* __restrict__ cellMinV, const int* __restrict__ cellMinI,
                        float* __restrict__ out) {
    const int m = blockIdx.x, tid = threadIdx.x;   // 32 blocks x 256 threads
    __shared__ float mv[256]; __shared__ int mi[256];
    __shared__ float kv[256]; __shared__ int ki[256];

    // ---- bg: global min over the upsampled map (reduced across cells) ----
    mv[tid] = cellMinV[m * 256 + tid];
    mi[tid] = cellMinI[m * 256 + tid];
    __syncthreads();
    for (int off = 128; off > 0; off >>= 1) {
        if (tid < off) {
            const float v2 = mv[tid + off]; const int i2 = mi[tid + off];
            if (v2 < mv[tid] || (v2 == mv[tid] && i2 < mi[tid])) { mv[tid] = v2; mi[tid] = i2; }
        }
        __syncthreads();
    }
    if (tid == 0) {
        const int f = mi[0];
        out[NREF * 256 * 3 + m * 2 + 0] = (float)(f % IMG);  // x = col
        out[NREF * 256 * 3 + m * 2 + 1] = (float)(f / IMG);  // y = row
    }

    // ---- stable descending sort of masked scores ----
    const float sc0 = cellS[m * 256 + tid];
    kv[tid] = (sc0 > THRESH) ? sc0 : -1.0f;
    ki[tid] = tid;
    __syncthreads();
    for (int k = 2; k <= 256; k <<= 1) {
        for (int j = k >> 1; j > 0; j >>= 1) {
            const int ixj = tid ^ j;
            if (ixj > tid) {
                const float a = kv[tid];  const int ia = ki[tid];
                const float c = kv[ixj];  const int ic = ki[ixj];
                const bool aGreater = (a > c) || (a == c && ia < ic);
                const bool descRegion = ((tid & k) == 0);
                const bool doSwap = descRegion ? !aGreater : aGreater;
                if (doSwap) { kv[tid] = c; ki[tid] = ic; kv[ixj] = a; ki[ixj] = ia; }
            }
            __syncthreads();
        }
    }
    const float sc = kv[tid];
    const int cid  = ki[tid];
    float x, y, s3;
    if (sc > THRESH) { x = cellX[m * 256 + cid]; y = cellY[m * 256 + cid]; s3 = sc; }
    else             { x = -1.0f; y = -1.0f; s3 = -1.0f; }
    float* rowp = out + (m * 256 + tid) * 3;
    rowp[0] = x; rowp[1] = y; rowp[2] = s3;
}

// ---------------------------------------------------------------------------
extern "C" void kernel_launch(void* const* d_in, const int* in_sizes, int n_in,
                              void* d_out, int out_size, void* d_ws, size_t ws_size,
                              hipStream_t stream) {
    const float* feat = (const float*)d_in[0];  // [1,256,64,64]
    const float* ref  = (const float*)d_in[1];  // [32,1,256]
    float* out = (float*)d_out;                 // 24576 + 64 floats
    char* ws = (char*)d_ws;

    _Float16* refA  = (_Float16*)(ws);                       //   16384 B
    _Float16* featB = (_Float16*)(ws + 16384);               // 2097152 B
    float* sim      = (float*)(ws + 2113536);                //  524288 B
    float* cellS    = (float*)(ws + 2637824);                //   32768 B
    float* cellX    = (float*)(ws + 2670592);                //   32768 B
    float* cellY    = (float*)(ws + 2703360);                //   32768 B
    float* cellMinV = (float*)(ws + 2736128);                //   32768 B
    int*   cellMinI = (int*)  (ws + 2768896);                //   32768 B

    k_ref_pack <<<NREF, 64, 0, stream>>>(ref, refA);
    k_feat_pack<<<16, 256, 0, stream>>>(feat, featB);
    k_gemm     <<<64, 128, 0, stream>>>(refA, featB, sim);   // 256 waves, 16 wmma each
    k_cells    <<<NREF * 256, 256, 0, stream>>>(sim, cellS, cellX, cellY, cellMinV, cellMinI);
    k_final    <<<NREF, 256, 0, stream>>>(cellS, cellX, cellY, cellMinV, cellMinI, out);
}
